// TextRNNRegressor_74062416052999
// MI455X (gfx1250) — matrix operand order repaired and checked by reading, not compiled, not run on hardware
//
#include <hip/hip_runtime.h>
#include <hip/hip_bf16.h>
#include <math.h>

typedef __attribute__((ext_vector_type(16))) _Float16 v16h;
typedef __attribute__((ext_vector_type(8)))  float    v8f;

namespace {
constexpr int kVocab = 512;
constexpr int kEmb   = 128;
constexpr int kHid   = 64;
constexpr int kT     = 1024;
constexpr int kHSZ   = 16 * kHid;                      // halves per h buffer

// ---- dynamic LDS layout (bytes), all 16B aligned ----
constexpr int kOffEmb = 0;                             // 512*128 f16   = 131072
constexpr int kOffX   = kOffEmb + kVocab * kEmb * 2;   // 16*1024 u16   =  32768
constexpr int kOffH0  = kOffX   + 16 * kT * 2;         // 2*[16][64]f16 =   4096
constexpr int kOffH1  = kOffH0  + 2 * kHSZ * 2;        // 2*[16][64]f16 =   4096
constexpr int kOffHF  = kOffH1  + 2 * kHSZ * 2;        // [16][64] f32  =   4096
constexpr int kSmemBytes = kOffHF + 16 * kHid * 4;     // 176128 bytes
} // namespace

// Compiler-level memory scheduling fence: pins LDS ops into the barrier
// signal->wait window (atomic fences alone allow sinking past the wait).
#define COMPILER_MEM_BARRIER() asm volatile("" ::: "memory")

// ---- split workgroup barrier: signal, overlap independent work, then wait ----
#if __has_builtin(__builtin_amdgcn_s_barrier_signal) && __has_builtin(__builtin_amdgcn_s_barrier_wait)
#define WG_BARRIER_SIGNAL()                                         \
    do {                                                            \
        __builtin_amdgcn_fence(__ATOMIC_RELEASE, "workgroup");      \
        __builtin_amdgcn_s_barrier_signal(-1);                      \
    } while (0)
#define WG_BARRIER_WAIT()                                           \
    do {                                                            \
        __builtin_amdgcn_s_barrier_wait(-1);                        \
        __builtin_amdgcn_fence(__ATOMIC_ACQUIRE, "workgroup");      \
    } while (0)
#else
#define WG_BARRIER_SIGNAL() __builtin_amdgcn_fence(__ATOMIC_RELEASE, "workgroup")
#define WG_BARRIER_WAIT()   __syncthreads()
#endif

union AFrag {
    v16h h;
    uint4 q[2];
};

// Load a 16x32 f16 A-fragment from an LDS half buffer.
// Per ISA: lane (m = l&15, hi = l>>4); halves j=0..7 -> K = 8*hi + j,
// halves j=8..15 -> K = 16 + 8*hi + (j-8)  => two 16B LDS loads.
__device__ inline v16h load_a_lds(const _Float16* __restrict__ base, int off0) {
    AFrag f;
    f.q[0] = *(const uint4*)(base + off0);
    f.q[1] = *(const uint4*)(base + off0 + 16);
    return f.h;
}

// Build a 32x16 f16 B-fragment (W^T chunk) from row-major f32 W[64][kdim].
// Dense-B striping: lane n = l&15; halves j=0..7 -> K = 8*hi + j, j=8..15 -> K = 16 + 8*hi + (j-8).
__device__ inline v16h load_b_weight(const float* __restrict__ W, int n, int kdim,
                                     int kbase, int hi) {
    v16h b;
#pragma unroll
    for (int j = 0; j < 16; ++j) b[j] = (_Float16)W[n * kdim + kbase + 8 * hi + j + (j >= 8 ? 8 : 0)];
    return b;
}

// tanh: native v_tanh_f32 on CDNA5 when available; else branch-free exp2 form.
__device__ inline float fast_tanh(float x) {
#if __has_builtin(__builtin_amdgcn_tanhf)
    return __builtin_amdgcn_tanhf(x);
#else
    float ax = __builtin_fabsf(x);
    float e  = __builtin_amdgcn_exp2f(ax * -2.8853900817779268f);  // exp(-2|x|)
    float r  = (1.f - e) * __builtin_amdgcn_rcpf(1.f + e);
    return __builtin_copysignf(r, x);
#endif
}

__global__ __launch_bounds__(128)
void TextRNNRegressor_kernel(const int* __restrict__ x,
                             const float* __restrict__ emb,
                             const float* __restrict__ Wih0, const float* __restrict__ Whh0,
                             const float* __restrict__ bih0, const float* __restrict__ bhh0,
                             const float* __restrict__ Wih1, const float* __restrict__ Whh1,
                             const float* __restrict__ bih1, const float* __restrict__ bhh1,
                             const float* __restrict__ W1, const float* __restrict__ b1,
                             const float* __restrict__ W2, const float* __restrict__ b2,
                             float* __restrict__ out) {
    extern __shared__ char smem[];
    _Float16*       sEmb = (_Float16*)(smem + kOffEmb);       // [512][128] f16
    unsigned short* sX   = (unsigned short*)(smem + kOffX);   // [16][1024] u16
    _Float16*       sH0  = (_Float16*)(smem + kOffH0);        // [2][16][64] f16
    _Float16*       sH1  = (_Float16*)(smem + kOffH1);        // [2][16][64] f16
    float*          sHF  = (float*)(smem + kOffHF);           // [16][64] f32

    const int tid  = threadIdx.x;
    const int lane = tid & 31;
    const int wv   = tid >> 5;           // 4 waves: each owns 16 hidden units
    const int b0   = blockIdx.x * 16;    // batch-tile base row
    const int m16  = lane & 15;
    const int hi   = lane >> 4;
    const int ng   = wv * 16 + m16;      // this lane's N column (global hidden idx)

    // ---------------- one-time LDS staging ----------------
    for (int i = tid; i < kVocab * kEmb; i += 128) sEmb[i] = (_Float16)emb[i];
    for (int i = tid; i < 16 * kT; i += 128) {
        int m = i >> 10, tt = i & (kT - 1);
        sX[i] = (unsigned short)x[(b0 + m) * kT + tt];
    }
    for (int i = tid; i < 2 * kHSZ; i += 128) {
        sH0[i] = (_Float16)0.f;
        sH1[i] = (_Float16)0.f;
    }

    // -------- weights -> register B-fragments (loop invariant, ~80 VGPRs) --------
    v16h bWih0[4], bWhh0[2], bWih1[2], bWhh1[2];
#pragma unroll
    for (int c = 0; c < 4; ++c) bWih0[c] = load_b_weight(Wih0, ng, kEmb, c * 32, hi);
#pragma unroll
    for (int c = 0; c < 2; ++c) bWhh0[c] = load_b_weight(Whh0, ng, kHid, c * 32, hi);
#pragma unroll
    for (int c = 0; c < 2; ++c) bWih1[c] = load_b_weight(Wih1, ng, kHid, c * 32, hi);
#pragma unroll
    for (int c = 0; c < 2; ++c) bWhh1[c] = load_b_weight(Whh1, ng, kHid, c * 32, hi);

    const float bias0 = bih0[ng] + bhh0[ng];
    const float bias1 = bih1[ng] + bhh1[ng];
    const int   hOff   = m16 * kHid + 8 * hi;   // base A-offset into an h buffer
    const int   dRow   = 8 * hi;                // D layout: VGPR r -> M = r + 8*hi
    const int   hStore = dRow * kHid + ng;      // store base: + r*kHid per accum row
    const int   aEOff  = 8 * hi;                // emb A-fragment lane offset

    __syncthreads();

    // ================ software-pipelined recurrence ================
    // Iteration t computes layer0(t) AND layer1(t-1); both read the already-
    // published h0(t-1): one barrier per step, shared h0 A-fragments.
    // Split barrier: publish h -> signal -> [pinned] load next step's emb
    // fragments + t+2 x-index -> wait. Compiler memory barriers keep those
    // loads inside the signal/wait window so they fly during barrier skew.

    v16h aE[4];     // loop-carried: embedding A-fragments for the upcoming step
    int  erowNext;  // loop-carried: emb row offset for the step after that

    // ---- prologue: layer0(0) (h0(-1) == 0 buffer) ----
    {
        const int erow0 = ((int)sX[m16 * kT + 0]) * kEmb;
#pragma unroll
        for (int c = 0; c < 4; ++c) aE[c] = load_a_lds(sEmb, erow0 + c * 32 + aEOff);

        v8f acc0;
#pragma unroll
        for (int r = 0; r < 8; ++r) acc0[r] = bias0;
#pragma unroll
        for (int c = 0; c < 4; ++c)
            acc0 = __builtin_amdgcn_wmma_f32_16x16x32_f16(false, aE[c], false, bWih0[c],
                                                          (short)0, acc0, false, false);
        _Float16* h0w = sH0 + 1 * kHSZ;   // parity for t=0: write buf 1
#pragma unroll
        for (int r = 0; r < 8; ++r)
            h0w[hStore + r * kHid] = (_Float16)fast_tanh(acc0[r]);
    }
    WG_BARRIER_SIGNAL();
    COMPILER_MEM_BARRIER();
    {   // overlap: step-1 emb fragments + prefetch step-2 index
        const int erow1 = ((int)sX[m16 * kT + 1]) * kEmb;
#pragma unroll
        for (int c = 0; c < 4; ++c) aE[c] = load_a_lds(sEmb, erow1 + c * 32 + aEOff);
        erowNext = ((int)sX[m16 * kT + 2]) * kEmb;
    }
    COMPILER_MEM_BARRIER();
    WG_BARRIER_WAIT();

    // ---- main loop: t = 1 .. T-1 (unroll 2 => compile-time buffer parity) ----
#pragma unroll 2
    for (int t = 1; t < kT; ++t) {
        const int p = t & 1;
        const _Float16* h0r = sH0 + p * kHSZ;        // h0(t-1)
        _Float16*       h0w = sH0 + (p ^ 1) * kHSZ;  // h0(t)
        const _Float16* h1r = sH1 + p * kHSZ;        // h1(t-2)
        _Float16*       h1w = sH1 + (p ^ 1) * kHSZ;  // h1(t-1)

        // --- h-state fragment loads (aE already in registers) ---
        v16h aH[2], aR[2];
#pragma unroll
        for (int c = 0; c < 2; ++c) aH[c] = load_a_lds(h0r, hOff + c * 32);
#pragma unroll
        for (int c = 0; c < 2; ++c) aR[c] = load_a_lds(h1r, hOff + c * 32);

        // --- layer0(t): x_t@Wih0^T + h0(t-1)@Whh0^T + bias0 ---
        v8f acc0;
#pragma unroll
        for (int r = 0; r < 8; ++r) acc0[r] = bias0;
#pragma unroll
        for (int c = 0; c < 4; ++c)
            acc0 = __builtin_amdgcn_wmma_f32_16x16x32_f16(false, aE[c], false, bWih0[c],
                                                          (short)0, acc0, false, false);
#pragma unroll
        for (int c = 0; c < 2; ++c)
            acc0 = __builtin_amdgcn_wmma_f32_16x16x32_f16(false, aH[c], false, bWhh0[c],
                                                          (short)0, acc0, false, false);

        // --- layer1(t-1): h0(t-1)@Wih1^T + h1(t-2)@Whh1^T + bias1 (independent chain) ---
        v8f acc1;
#pragma unroll
        for (int r = 0; r < 8; ++r) acc1[r] = bias1;
#pragma unroll
        for (int c = 0; c < 2; ++c)
            acc1 = __builtin_amdgcn_wmma_f32_16x16x32_f16(false, aH[c], false, bWih1[c],
                                                          (short)0, acc1, false, false);
#pragma unroll
        for (int c = 0; c < 2; ++c)
            acc1 = __builtin_amdgcn_wmma_f32_16x16x32_f16(false, aR[c], false, bWhh1[c],
                                                          (short)0, acc1, false, false);

        // --- publish h0(t) and h1(t-1): fixed base + r*kHid immediates ---
#pragma unroll
        for (int r = 0; r < 8; ++r) {
            h0w[hStore + r * kHid] = (_Float16)fast_tanh(acc0[r]);
            h1w[hStore + r * kHid] = (_Float16)fast_tanh(acc1[r]);
        }

        WG_BARRIER_SIGNAL();
        COMPILER_MEM_BARRIER();
        // pinned overlap: emb fragments for t+1 (address ready two steps back)
        // + issue the t+2 index load (consumed next iteration)
#pragma unroll
        for (int c = 0; c < 4; ++c) aE[c] = load_a_lds(sEmb, erowNext + c * 32 + aEOff);
        const int tnn = (t + 2 < kT) ? (t + 2) : (kT - 1);
        erowNext = ((int)sX[m16 * kT + tnn]) * kEmb;
        COMPILER_MEM_BARRIER();
        WG_BARRIER_WAIT();
    }

    // ---- epilogue: layer1(T-1) from h0(T-1), h1(T-2) (both in parity-0 buffers) ----
    {
        const _Float16* h0r = sH0 + 0 * kHSZ;
        const _Float16* h1r = sH1 + 0 * kHSZ;
        v16h aH[2], aR[2];
#pragma unroll
        for (int c = 0; c < 2; ++c) aH[c] = load_a_lds(h0r, hOff + c * 32);
#pragma unroll
        for (int c = 0; c < 2; ++c) aR[c] = load_a_lds(h1r, hOff + c * 32);

        v8f acc1;
#pragma unroll
        for (int r = 0; r < 8; ++r) acc1[r] = bias1;
#pragma unroll
        for (int c = 0; c < 2; ++c)
            acc1 = __builtin_amdgcn_wmma_f32_16x16x32_f16(false, aH[c], false, bWih1[c],
                                                          (short)0, acc1, false, false);
#pragma unroll
        for (int c = 0; c < 2; ++c)
            acc1 = __builtin_amdgcn_wmma_f32_16x16x32_f16(false, aR[c], false, bWhh1[c],
                                                          (short)0, acc1, false, false);

        // final hidden h_n in f32
#pragma unroll
        for (int r = 0; r < 8; ++r)
            sHF[(r + dRow) * kHid + ng] = fast_tanh(acc1[r]);
    }
    __syncthreads();

    // ---------------- MLP head: y = relu(h @ W1^T + b1) @ W2^T + b2 ----------------
    if (tid < 16) {
        const float* hrow = sHF + tid * kHid;
        float y = b2[0];
#pragma unroll 4
        for (int j = 0; j < 32; ++j) {
            float z = b1[j];
            const float* w = W1 + j * kHid;
#pragma unroll 8
            for (int k = 0; k < kHid; ++k) z += hrow[k] * w[k];
            y += fmaxf(z, 0.f) * W2[j];
        }
        out[b0 + tid] = y;
    }
}

extern "C" void kernel_launch(void* const* d_in, const int* in_sizes, int n_in,
                              void* d_out, int out_size, void* d_ws, size_t ws_size,
                              hipStream_t stream) {
    (void)in_sizes; (void)n_in; (void)out_size; (void)d_ws; (void)ws_size;

    const int*   x    = (const int*)d_in[0];
    const float* emb  = (const float*)d_in[1];
    const float* Wih0 = (const float*)d_in[2];
    const float* Whh0 = (const float*)d_in[3];
    const float* bih0 = (const float*)d_in[4];
    const float* bhh0 = (const float*)d_in[5];
    const float* Wih1 = (const float*)d_in[6];
    const float* Whh1 = (const float*)d_in[7];
    const float* bih1 = (const float*)d_in[8];
    const float* bhh1 = (const float*)d_in[9];
    const float* W1   = (const float*)d_in[10];
    const float* b1   = (const float*)d_in[11];
    const float* W2   = (const float*)d_in[12];
    const float* b2   = (const float*)d_in[13];
    float* out = (float*)d_out;

    // Allow >64KB dynamic LDS (CDNA5 supports up to 320KB per workgroup).
    hipFuncSetAttribute((const void*)TextRNNRegressor_kernel,
                        hipFuncAttributeMaxDynamicSharedMemorySize, kSmemBytes);

    // 16 batch tiles of 16 rows; 4 waves (128 threads) per tile.
    TextRNNRegressor_kernel<<<dim3(16), dim3(128), kSmemBytes, stream>>>(
        x, emb, Wih0, Whh0, bih0, bhh0, Wih1, Whh1, bih1, bhh1, W1, b1, W2, b2, out);
}
